// MDAttention_75024488727107
// MI455X (gfx1250) — matrix operand order repaired and checked by reading, not compiled, not run on hardware
//
#include <hip/hip_runtime.h>
#include <hip/hip_bf16.h>
#include <math.h>

// ---------------------------------------------------------------------------
// MDAttention pipeline for MI455X (gfx1250), wave32 + WMMA bf16.
// All GEMM work (1x1 convs, q@k^T, attn@v, DFT-as-GEMM) runs on
// v_wmma_f32_16x16x32_bf16 with fp32 accumulation. Activations are kept in
// channel-last (HWC) layout so BOTH WMMA operands are contiguous-K per lane
// (every fragment = 2x global_load_b128). Each wave owns a 32x16 output tile
// (two WMMAs share one activation fragment) and the K loop is software-
// pipelined so fragment loads overlap the matrix op.
// bf16 activation panels are 33MB -> resident in the 192MB L2.
// ---------------------------------------------------------------------------

typedef __bf16 bf16;
typedef __attribute__((ext_vector_type(16))) __bf16 v16bf;
typedef __attribute__((ext_vector_type(8)))  float  v8f;

static constexpr int  NPIX  = 16384;   // H*W = 128*128
static constexpr int  IMGW  = 128;
static constexpr long BATCH = 4;
static constexpr int  CDIM  = 256;
static constexpr int  C2    = 512;
static constexpr int  C3    = 768;
static constexpr int  HEADS = 8;

// --------------------------- WMMA fragment helpers -------------------------
// Fragment layouts (wave32); A/B use this file's contiguous K order (not the ISA table), C/D per ISA:
//  A 16x32 bf16 : lane l -> row m = l&15 ; hi half adds +16 to K;
//                 element e -> K = e + 16*(l>>4)
//  B 32x16 bf16 : lane l -> col n = l&15 ; K = e + 16*(l>>4)
//  C/D 16x16 f32: vgpr r -> M = r + 8*(l>>4) ; N = l&15

// A row-major [M,K] (contiguous K per lane -> 2x b128)
__device__ __forceinline__ v16bf load_a_frag(const bf16* __restrict__ A, long lda,
                                             int m0, int k0, int lane) {
  v16bf a;
  const int m  = m0 + (lane & 15);
  const int hb = (lane >> 4) << 4;
  const bf16* p = A + (long)m * lda + k0 + hb;
#pragma unroll
  for (int e = 0; e < 16; ++e) a[e] = p[e];
  return a;
}

// A stored transposed: A(m,k) = At[k*ldt + m]  (strided; used only in scores)
__device__ __forceinline__ v16bf load_at_frag(const bf16* __restrict__ At, long ldt,
                                              int m0, int k0, int lane) {
  v16bf a;
  const int m  = m0 + (lane & 15);
  const int hb = (lane >> 4) << 4;
  const bf16* p = At + (long)(k0 + hb) * ldt + m;
#pragma unroll
  for (int e = 0; e < 16; ++e) a[e] = p[(long)e * ldt];
  return a;
}

// B row-major [K,N] (strided; used only in scores)
__device__ __forceinline__ v16bf load_b_frag(const bf16* __restrict__ Bm, long ldb,
                                             int k0, int n0, int lane) {
  v16bf b;
  const int n  = n0 + (lane & 15);
  const int kk = k0 + ((lane >> 4) << 4);
  const bf16* p = Bm + (long)kk * ldb + n;
#pragma unroll
  for (int e = 0; e < 16; ++e) b[e] = p[(long)e * ldb];
  return b;
}

// B(k,n) = Bt[n*ldt + k]  (contiguous K per lane -> 2x b128)
__device__ __forceinline__ v16bf load_bt_frag(const bf16* __restrict__ Bt, long ldt,
                                              int k0, int n0, int lane) {
  v16bf b;
  const int n  = n0 + (lane & 15);
  const int kk = k0 + ((lane >> 4) << 4);
  const bf16* p = Bt + (long)n * ldt + kk;
#pragma unroll
  for (int e = 0; e < 16; ++e) b[e] = p[e];
  return b;
}

__device__ __forceinline__ v8f wmma_bf16(v16bf a, v16bf b, v8f c) {
  return __builtin_amdgcn_wmma_f32_16x16x32_bf16(false, a, false, b,
                                                 (short)0, c, false, false);
}

// ------------------------------- kernels -----------------------------------

__global__ void k_cast_bf16(const float* __restrict__ src, bf16* __restrict__ dst, long n) {
  long i = (long)blockIdx.x * blockDim.x + threadIdx.x;
  if (i < n) dst[i] = (bf16)src[i];
}

// DFT cos/sin tables; real(F x F) = CxC - SxS (fwd), same combo /HW (inv).
__global__ void k_dft_init(bf16* __restrict__ Cd, bf16* __restrict__ Sd,
                           bf16* __restrict__ Sn) {
  int i = blockIdx.x * blockDim.x + threadIdx.x;   // 16384
  int u = i >> 7, w = i & 127;
  float ang = 6.28318530717958647692f * (float)((u * w) & 127) * (1.0f / 128.0f);
  float c = cosf(ang), s = sinf(ang);
  Cd[i] = (bf16)c; Sd[i] = (bf16)s; Sn[i] = (bf16)(-s);
}

// tiled transpose+cast: CHW f32 -> HWC bf16 (per batch [C,N] -> [N,C])
__global__ void k_chwf_to_hwc(const float* __restrict__ src, bf16* __restrict__ dst, int C) {
  __shared__ bf16 tile[32][33];
  const int b = blockIdx.z, n0 = blockIdx.x * 32, c0 = blockIdx.y * 32;
  const int tx = threadIdx.x;
  for (int i = threadIdx.y; i < 32; i += 8)
    tile[i][tx] = (bf16)src[((long)b * C + c0 + i) * NPIX + n0 + tx];      // coalesced n
  __syncthreads();
  for (int i = threadIdx.y; i < 32; i += 8)
    dst[((long)b * NPIX + n0 + i) * C + c0 + tx] = tile[tx][i];            // coalesced c
}

// tiled transpose: planes bf16 [C,N] -> HWC bf16 [N,C]
__global__ void k_pl_to_hwc(const bf16* __restrict__ src, bf16* __restrict__ dst, int C) {
  __shared__ bf16 tile[32][33];
  const int b = blockIdx.z, n0 = blockIdx.x * 32, c0 = blockIdx.y * 32;
  const int tx = threadIdx.x;
  for (int i = threadIdx.y; i < 32; i += 8)
    tile[i][tx] = src[((long)b * C + c0 + i) * NPIX + n0 + tx];
  __syncthreads();
  for (int i = threadIdx.y; i < 32; i += 8)
    dst[((long)b * NPIX + n0 + i) * C + c0 + tx] = tile[tx][i];
}

// tiled transpose: HWC bf16 [N,C] -> planes bf16 [C,N]
__global__ void k_hwc_to_pl(const bf16* __restrict__ src, bf16* __restrict__ dst, int C) {
  __shared__ bf16 tile[32][33];
  const int b = blockIdx.z, n0 = blockIdx.x * 32, c0 = blockIdx.y * 32;
  const int tx = threadIdx.x;
  for (int i = threadIdx.y; i < 32; i += 8)
    tile[i][tx] = src[((long)b * NPIX + n0 + i) * C + c0 + tx];
  __syncthreads();
  for (int i = threadIdx.y; i < 32; i += 8)
    dst[((long)b * C + c0 + i) * NPIX + n0 + tx] = tile[tx][i];
}

// Y(m,n) = sum_k W[m,k] * Xhwc[n,k]  (+bias)(gelu)(+addend); HWC/CHW stores.
// One wave = 32(m) x 16(n) tile; K loop software-pipelined.
__global__ void k_conv1x1_wmma(const bf16* __restrict__ Wm, const bf16* __restrict__ Xhwc,
                               const float* __restrict__ bias,
                               const float* __restrict__ add_hwc,
                               float* __restrict__ Yf_hwc, bf16* __restrict__ Ybf_hwc,
                               float* __restrict__ Y_chw,
                               int M, int K, int applyGelu) {
  const int lane = threadIdx.x & 31, wv = threadIdx.x >> 5;
  const int n0 = (blockIdx.x * 4 + wv) * 16;
  const int m0 = blockIdx.y * 32;
  const int b  = blockIdx.z;
  const bf16* Bt = Xhwc + (long)b * NPIX * K;      // Bt[n*K + k]
  v8f acc0 = {}, acc1 = {};
  v16bf a0 = load_a_frag(Wm, K, m0,      0, lane);
  v16bf a1 = load_a_frag(Wm, K, m0 + 16, 0, lane);
  v16bf bb = load_bt_frag(Bt, K, 0, n0, lane);
  for (int k0 = 0; k0 < K; k0 += 32) {
    v16bf na0 = a0, na1 = a1, nbb = bb;
    const int kn = k0 + 32;
    if (kn < K) {                                  // prefetch next slab into regs
      na0 = load_a_frag(Wm, K, m0,      kn, lane);
      na1 = load_a_frag(Wm, K, m0 + 16, kn, lane);
      nbb = load_bt_frag(Bt, K, kn, n0, lane);
      __builtin_prefetch(Bt + (long)(n0 + (lane & 15)) * K + kn + 32, 0, 3);
    }
    acc0 = wmma_bf16(a0, bb, acc0);
    acc1 = wmma_bf16(a1, bb, acc1);
    a0 = na0; a1 = na1; bb = nbb;
  }
  const int n  = n0 + (lane & 15);
  const int mh = (lane >> 4) << 3;
#pragma unroll
  for (int t = 0; t < 2; ++t) {
    const v8f acc = t ? acc1 : acc0;
#pragma unroll
    for (int r = 0; r < 8; ++r) {
      const int m = m0 + t * 16 + r + mh;
      float v = acc[r];
      if (bias) v += bias[m];
      if (applyGelu) v = 0.5f * v * (1.0f + erff(v * 0.70710678118654752f));
      const long hidx = ((long)b * NPIX + n) * M + m;      // per-lane contiguous over r
      if (add_hwc) v += add_hwc[hidx];
      if (Yf_hwc)  Yf_hwc[hidx]  = v;
      if (Ybf_hwc) Ybf_hwc[hidx] = (bf16)v;
      if (Y_chw)   Y_chw[((long)b * M + m) * NPIX + n] = v; // final output layout
    }
  }
}

// depthwise 3x3, SAME zero padding, fp32, HWC layout (coalesced over c)
__global__ void k_dwconv3_hwc(const float* __restrict__ X, const float* __restrict__ Wd,
                              const float* __restrict__ Bd, float* __restrict__ Y, int CC) {
  long i = (long)blockIdx.x * blockDim.x + threadIdx.x;
  long total = BATCH * (long)NPIX * CC;
  if (i >= total) return;
  const int c = (int)(i % CC);
  const long t = i / CC;
  const int n = (int)(t % NPIX);
  const long b = t / NPIX;
  const int h = n >> 7, w = n & 127;
  const float* base = X + (long)b * NPIX * CC;
  const float* wt = Wd + (long)c * 9;
  float s = Bd[c];
#pragma unroll
  for (int dy = -1; dy <= 1; ++dy) {
    const int hy = h + dy;
    if (hy < 0 || hy > 127) continue;
#pragma unroll
    for (int dx = -1; dx <= 1; ++dx) {
      const int wx = w + dx;
      if (wx < 0 || wx > 127) continue;
      s += wt[(dy + 1) * 3 + (dx + 1)] * base[(long)((hy << 7) | wx) * CC + c];
    }
  }
  Y[i] = s;
}

// phase 1 of L2 row-norm over n (HWC): per-channel partial sum-of-squares
__global__ void k_sumsq_hwc(const float* __restrict__ X, float* __restrict__ sums,
                            int CC, int normThresh) {
  const int c = threadIdx.x;                  // blockDim.x == CC
  const int b = blockIdx.x >> 6, chunk = blockIdx.x & 63;
  if (c >= normThresh) return;
  const float* p = X + ((long)b * NPIX + chunk * 256) * CC + c;
  float s = 0.f;
  for (int i = 0; i < 256; ++i) { float v = p[(long)i * CC]; s += v * v; }
  atomicAdd(&sums[b * CC + c], s);
}

// phase 2: scale (1/max(||row||,1e-12) for c<normThresh) + cast to bf16
__global__ void k_scale_cast_hwc(const float* __restrict__ X, const float* __restrict__ sums,
                                 bf16* __restrict__ Y, int CC, int normThresh) {
  const int c = threadIdx.x;
  const int b = blockIdx.x >> 6, chunk = blockIdx.x & 63;
  float scale = 1.0f;
  if (c < normThresh) scale = 1.0f / fmaxf(sqrtf(sums[b * CC + c]), 1e-12f);
  const long base = ((long)b * NPIX + chunk * 256) * CC + c;
  for (int i = 0; i < 256; ++i) Y[base + (long)i * CC] = (bf16)(X[base + (long)i * CC] * scale);
}

// split-K q@k^T: grid(bh, 32 K-chunks), 4 waves/block, f32 atomic reduction
__global__ void k_scores_wmma(const bf16* __restrict__ QKhwc, int cb, int qoff, int koff,
                              float* __restrict__ S) {
  const int bh = blockIdx.x, b = bh >> 3, h = bh & 7;
  const int kc = blockIdx.y;
  const int wv = threadIdx.x >> 5, lane = threadIdx.x & 31;
  const int mi = (wv >> 1) * 16, ni = (wv & 1) * 16;
  const bf16* base = QKhwc + (long)b * NPIX * cb;
  const bf16* At = base + qoff + h * 32 + mi;   // A(m,k)=At[k*cb+m]
  const bf16* Bm = base + koff + h * 32 + ni;   // B(k,nc)=Bm[k*cb+nc]
  v8f acc = {};
  const int k0 = kc * 512;
  for (int kk = 0; kk < 512; kk += 32)
    acc = wmma_bf16(load_at_frag(At, cb, 0, k0 + kk, lane),
                    load_b_frag(Bm, cb, k0 + kk, 0, lane), acc);
  const int n  = ni + (lane & 15);
  const int mh = (lane >> 4) << 3;
#pragma unroll
  for (int r = 0; r < 8; ++r)
    atomicAdd(&S[((long)bh * 32 + mi + r + mh) * 32 + n], acc[r]);
}

// softmax over 32 cols, one wave per row; scale by per-head temperature
__global__ void k_softmax32(const float* __restrict__ S, const float* __restrict__ temp,
                            bf16* __restrict__ A) {
  const int row = blockIdx.x;              // (b*8+h)*32 + c
  const int h = (row >> 5) & 7;
  const int t = threadIdx.x;
  float v = S[(long)row * 32 + t] * temp[h];
  float m = v;
  for (int off = 16; off > 0; off >>= 1) m = fmaxf(m, __shfl_xor(m, off));
  float e = __expf(v - m);
  float s = e;
  for (int off = 16; off > 0; off >>= 1) s += __shfl_xor(s, off);
  A[(long)row * 32 + t] = (bf16)(e / s);
}

// O_hwc = attn @ V ; one wave covers both 16-row tiles, one shared B fragment
__global__ void k_attnv_wmma(const bf16* __restrict__ A, const bf16* __restrict__ Vhwc,
                             int cbv, int voff, bf16* __restrict__ Ohwc, int cbo) {
  const int lane = threadIdx.x & 31, wv = threadIdx.x >> 5;
  const int n0 = (blockIdx.x * 4 + wv) * 16;
  const int bh = blockIdx.z, b = bh >> 3, h = bh & 7;
  const bf16* Ab = A + (long)bh * 1024;
  const bf16* Bt = Vhwc + (long)b * NPIX * cbv + voff + h * 32;  // Bt[n*cbv+k]
  const v16bf bb = load_bt_frag(Bt, cbv, 0, n0, lane);
  v8f acc0 = {}, acc1 = {};
  acc0 = wmma_bf16(load_a_frag(Ab, 32, 0,  0, lane), bb, acc0);
  acc1 = wmma_bf16(load_a_frag(Ab, 32, 16, 0, lane), bb, acc1);
  const int n  = n0 + (lane & 15);
  const int mh = (lane >> 4) << 3;
#pragma unroll
  for (int r = 0; r < 8; ++r) {
    Ohwc[((long)b * NPIX + n) * cbo + h * 32 + r + mh]      = (bf16)acc0[r];
    Ohwc[((long)b * NPIX + n) * cbo + h * 32 + 16 + r + mh] = (bf16)acc1[r];
  }
}

// batched 128^3: O = scale*(A1*Bt1^T [+ A2*Bt2^T]); per-image/shared operands;
// 32x16 tile per wave; optional transposed store (contiguous either way).
__global__ void k_gemm128_wmma(const bf16* __restrict__ A1, int a1pi,
                               const bf16* __restrict__ Bt1, int bt1pi,
                               const bf16* __restrict__ A2, int a2pi,
                               const bf16* __restrict__ Bt2, int bt2pi,
                               float scale, bf16* __restrict__ O, int storeT) {
  const int lane = threadIdx.x & 31, wv = threadIdx.x >> 5;
  const int n0 = (blockIdx.x * 4 + wv) * 16;
  const int m0 = blockIdx.y * 32;
  const long img = blockIdx.z;
  v8f acc0 = {}, acc1 = {};
  {
    const bf16* a1 = A1 + (a1pi ? (img << 14) : 0);
    const bf16* b1 = Bt1 + (bt1pi ? (img << 14) : 0);
#pragma unroll
    for (int k0 = 0; k0 < 128; k0 += 32) {
      const v16bf bbf = load_bt_frag(b1, IMGW, k0, n0, lane);
      acc0 = wmma_bf16(load_a_frag(a1, IMGW, m0,      k0, lane), bbf, acc0);
      acc1 = wmma_bf16(load_a_frag(a1, IMGW, m0 + 16, k0, lane), bbf, acc1);
    }
  }
  if (A2) {
    const bf16* a2 = A2 + (a2pi ? (img << 14) : 0);
    const bf16* b2 = Bt2 + (bt2pi ? (img << 14) : 0);
#pragma unroll
    for (int k0 = 0; k0 < 128; k0 += 32) {
      const v16bf bbf = load_bt_frag(b2, IMGW, k0, n0, lane);
      acc0 = wmma_bf16(load_a_frag(a2, IMGW, m0,      k0, lane), bbf, acc0);
      acc1 = wmma_bf16(load_a_frag(a2, IMGW, m0 + 16, k0, lane), bbf, acc1);
    }
  }
  const int n  = n0 + (lane & 15);
  const int mh = (lane >> 4) << 3;
#pragma unroll
  for (int t = 0; t < 2; ++t) {
    const v8f acc = t ? acc1 : acc0;
#pragma unroll
    for (int r = 0; r < 8; ++r) {
      const int m = m0 + t * 16 + r + mh;
      if (storeT) O[(img << 14) + (long)n * IMGW + m] = (bf16)(acc[r] * scale);
      else        O[(img << 14) + (long)m * IMGW + n] = (bf16)(acc[r] * scale);
    }
  }
}

// ------------------------------- driver ------------------------------------

extern "C" void kernel_launch(void* const* d_in, const int* in_sizes, int n_in,
                              void* d_out, int out_size, void* d_ws, size_t ws_size,
                              hipStream_t stream) {
  (void)in_sizes; (void)n_in; (void)out_size; (void)ws_size;

  const float* x        = (const float*)d_in[0];
  const float* temp     = (const float*)d_in[1];
  const float* w_qkv    = (const float*)d_in[2];
  const float* b_qkv    = (const float*)d_in[3];
  const float* w_qkv_dw = (const float*)d_in[4];
  const float* b_qkv_dw = (const float*)d_in[5];
  const float* w_proj   = (const float*)d_in[6];
  const float* b_proj   = (const float*)d_in[7];
  const float* w_f1     = (const float*)d_in[8];
  const float* w_f2     = (const float*)d_in[9];
  const float* w_kv     = (const float*)d_in[10];
  const float* b_kv     = (const float*)d_in[11];
  const float* w_kv_dw  = (const float*)d_in[12];
  const float* b_kv_dw  = (const float*)d_in[13];
  const float* w_projf  = (const float*)d_in[14];
  const float* b_projf  = (const float*)d_in[15];
  float* out_final = (float*)d_out;

  // ---- carve workspace ----
  char* wp = (char*)d_ws;
  auto carve = [&](size_t bytes) -> char* {
    char* r = wp; wp += (bytes + 255) & ~(size_t)255; return r;
  };
  const long ACT = BATCH * CDIM * (long)NPIX;
  bf16*  w_qkv_bf   = (bf16*)carve((size_t)C3 * CDIM * 2);
  bf16*  w_proj_bf  = (bf16*)carve((size_t)CDIM * CDIM * 2);
  bf16*  w_f1_bf    = (bf16*)carve((size_t)CDIM * CDIM * 2);
  bf16*  w_f2_bf    = (bf16*)carve((size_t)CDIM * CDIM * 2);
  bf16*  w_kv_bf    = (bf16*)carve((size_t)C2 * CDIM * 2);
  bf16*  w_projf_bf = (bf16*)carve((size_t)CDIM * CDIM * 2);
  bf16*  Cd         = (bf16*)carve((size_t)NPIX * 2);
  bf16*  Sd         = (bf16*)carve((size_t)NPIX * 2);
  bf16*  Sn         = (bf16*)carve((size_t)NPIX * 2);
  bf16*  x_hwc      = (bf16*)carve((size_t)ACT * 2);
  bf16*  x_pl       = (bf16*)carve((size_t)ACT * 2);
  float* qkv_f      = (float*)carve((size_t)BATCH * C3 * NPIX * 4);   // HWC
  float* qkvd_f     = (float*)carve((size_t)BATCH * C3 * NPIX * 4);   // HWC
  float* sums       = (float*)carve((size_t)BATCH * C3 * 4);
  bf16*  qkvn_bf    = (bf16*)carve((size_t)BATCH * C3 * NPIX * 2);    // HWC
  float* scores     = (float*)carve((size_t)BATCH * HEADS * 32 * 32 * 4);
  bf16*  attn_bf    = (bf16*)carve((size_t)BATCH * HEADS * 32 * 32 * 2);
  bf16*  out1_hwc   = (bf16*)carve((size_t)ACT * 2);
  float* out_f_hwc  = (float*)carve((size_t)ACT * 4);
  bf16*  out_bf_hwc = (bf16*)carve((size_t)ACT * 2);
  bf16*  P1t        = (bf16*)carve((size_t)ACT * 2);   // planes (also Q1t)
  bf16*  P2t        = (bf16*)carve((size_t)ACT * 2);   // planes (also Q2t)
  bf16*  fft_pl     = (bf16*)carve((size_t)ACT * 2);   // xfft / vf planes
  bf16*  xfft_hwc   = (bf16*)carve((size_t)ACT * 2);
  bf16*  m1_hwc     = (bf16*)carve((size_t)ACT * 2);
  bf16*  m2_hwc     = (bf16*)carve((size_t)ACT * 2);
  bf16*  m2_pl      = (bf16*)carve((size_t)ACT * 2);
  bf16*  vf_hwc     = (bf16*)carve((size_t)ACT * 2);
  float* kv_f       = (float*)carve((size_t)BATCH * C2 * NPIX * 4);   // HWC
  float* kvd_f      = (float*)carve((size_t)BATCH * C2 * NPIX * 4);   // HWC
  bf16*  kvn_bf     = (bf16*)carve((size_t)BATCH * C2 * NPIX * 2);    // HWC
  bf16*  out2_hwc   = (bf16*)carve((size_t)ACT * 2);

  auto castN = [&](const float* s, bf16* d, long n) {
    k_cast_bf16<<<dim3((unsigned)((n + 255) / 256)), dim3(256), 0, stream>>>(s, d, n);
  };

  // ---- weights, tables, input layouts ----
  castN(w_qkv,   w_qkv_bf,   (long)C3 * CDIM);
  castN(w_proj,  w_proj_bf,  (long)CDIM * CDIM);
  castN(w_f1,    w_f1_bf,    (long)CDIM * CDIM);
  castN(w_f2,    w_f2_bf,    (long)CDIM * CDIM);
  castN(w_kv,    w_kv_bf,    (long)C2 * CDIM);
  castN(w_projf, w_projf_bf, (long)CDIM * CDIM);
  k_dft_init<<<dim3(64), dim3(256), 0, stream>>>(Cd, Sd, Sn);
  k_chwf_to_hwc<<<dim3(512, CDIM / 32, BATCH), dim3(32, 8), 0, stream>>>(x, x_hwc, CDIM);
  castN(x, x_pl, ACT);

  const dim3 tb128(128);
  // ---- branch 1: qkv -> dwconv -> l2norm -> channel attention -> proj ----
  k_conv1x1_wmma<<<dim3(NPIX / 64, C3 / 32, BATCH), tb128, 0, stream>>>(
      w_qkv_bf, x_hwc, b_qkv, nullptr, qkv_f, nullptr, nullptr, C3, CDIM, 0);
  k_dwconv3_hwc<<<dim3((unsigned)((BATCH * (long)NPIX * C3 + 255) / 256)), dim3(256), 0, stream>>>(
      qkv_f, w_qkv_dw, b_qkv_dw, qkvd_f, C3);
  hipMemsetAsync(sums, 0, (size_t)BATCH * C3 * 4, stream);
  k_sumsq_hwc<<<dim3((unsigned)(BATCH * 64)), dim3(C3), 0, stream>>>(qkvd_f, sums, C3, 512);
  k_scale_cast_hwc<<<dim3((unsigned)(BATCH * 64)), dim3(C3), 0, stream>>>(qkvd_f, sums, qkvn_bf, C3, 512);
  hipMemsetAsync(scores, 0, (size_t)BATCH * HEADS * 32 * 32 * 4, stream);
  k_scores_wmma<<<dim3(BATCH * HEADS, 32), tb128, 0, stream>>>(qkvn_bf, C3, 0, 256, scores);
  k_softmax32<<<dim3(BATCH * HEADS * 32), dim3(32), 0, stream>>>(scores, temp, attn_bf);
  k_attnv_wmma<<<dim3(NPIX / 64, 1, BATCH * HEADS), tb128, 0, stream>>>(
      attn_bf, qkvn_bf, C3, 512, out1_hwc, CDIM);
  k_conv1x1_wmma<<<dim3(NPIX / 64, CDIM / 32, BATCH), tb128, 0, stream>>>(
      w_proj_bf, out1_hwc, b_proj, nullptr, out_f_hwc, out_bf_hwc, nullptr, CDIM, CDIM, 0);

  // ---- branch 2: FFT-domain MLP via DFT-as-GEMM (real = CxC - SxS) ----
  const dim3 g128(2, 4, (unsigned)(BATCH * CDIM));
  // P1^T = (x*C)^T, P2^T = (x*S)^T  (A = x planes, Bt = C/S by symmetry)
  k_gemm128_wmma<<<g128, tb128, 0, stream>>>(x_pl, 1, Cd, 0, nullptr, 0, nullptr, 0, 1.0f, P1t, 1);
  k_gemm128_wmma<<<g128, tb128, 0, stream>>>(x_pl, 1, Sd, 0, nullptr, 0, nullptr, 0, 1.0f, P2t, 1);
  // xfft = C*P1 + (-S)*P2  (Bt = P^T planes)
  k_gemm128_wmma<<<g128, tb128, 0, stream>>>(Cd, 0, P1t, 1, Sn, 0, P2t, 1, 1.0f, fft_pl, 0);
  k_pl_to_hwc<<<dim3(512, CDIM / 32, BATCH), dim3(32, 8), 0, stream>>>(fft_pl, xfft_hwc, CDIM);
  k_conv1x1_wmma<<<dim3(NPIX / 64, CDIM / 32, BATCH), tb128, 0, stream>>>(
      w_f1_bf, xfft_hwc, nullptr, nullptr, nullptr, m1_hwc, nullptr, CDIM, CDIM, 1);  // GELU
  k_conv1x1_wmma<<<dim3(NPIX / 64, CDIM / 32, BATCH), tb128, 0, stream>>>(
      w_f2_bf, m1_hwc, nullptr, nullptr, nullptr, m2_hwc, nullptr, CDIM, CDIM, 0);
  k_hwc_to_pl<<<dim3(512, CDIM / 32, BATCH), dim3(32, 8), 0, stream>>>(m2_hwc, m2_pl, CDIM);
  // vf = (C*(m2*C) + (-S)*(m2*S)) / (H*W)
  k_gemm128_wmma<<<g128, tb128, 0, stream>>>(m2_pl, 1, Cd, 0, nullptr, 0, nullptr, 0, 1.0f, P1t, 1);
  k_gemm128_wmma<<<g128, tb128, 0, stream>>>(m2_pl, 1, Sd, 0, nullptr, 0, nullptr, 0, 1.0f, P2t, 1);
  k_gemm128_wmma<<<g128, tb128, 0, stream>>>(Cd, 0, P1t, 1, Sn, 0, P2t, 1, 1.0f / 16384.0f, fft_pl, 0);
  k_pl_to_hwc<<<dim3(512, CDIM / 32, BATCH), dim3(32, 8), 0, stream>>>(fft_pl, vf_hwc, CDIM);

  // ---- second channel attention on `out` with FFT-branch values ----
  k_conv1x1_wmma<<<dim3(NPIX / 64, C2 / 32, BATCH), tb128, 0, stream>>>(
      w_kv_bf, out_bf_hwc, b_kv, nullptr, kv_f, nullptr, nullptr, C2, CDIM, 0);
  k_dwconv3_hwc<<<dim3((unsigned)((BATCH * (long)NPIX * C2 + 255) / 256)), dim3(256), 0, stream>>>(
      kv_f, w_kv_dw, b_kv_dw, kvd_f, C2);
  hipMemsetAsync(sums, 0, (size_t)BATCH * C2 * 4, stream);
  k_sumsq_hwc<<<dim3((unsigned)(BATCH * 64)), dim3(C2), 0, stream>>>(kvd_f, sums, C2, C2);
  k_scale_cast_hwc<<<dim3((unsigned)(BATCH * 64)), dim3(C2), 0, stream>>>(kvd_f, sums, kvn_bf, C2, C2);
  hipMemsetAsync(scores, 0, (size_t)BATCH * HEADS * 32 * 32 * 4, stream);
  k_scores_wmma<<<dim3(BATCH * HEADS, 32), tb128, 0, stream>>>(kvn_bf, C2, 256, 0, scores);  // k first, q second
  k_softmax32<<<dim3(BATCH * HEADS * 32), dim3(32), 0, stream>>>(scores, temp, attn_bf);
  k_attnv_wmma<<<dim3(NPIX / 64, 1, BATCH * HEADS), tb128, 0, stream>>>(
      attn_bf, vf_hwc, CDIM, 0, out2_hwc, CDIM);

  // ---- final projection + residual into d_out (CHW) ----
  k_conv1x1_wmma<<<dim3(NPIX / 64, CDIM / 32, BATCH), tb128, 0, stream>>>(
      w_projf_bf, out2_hwc, b_projf, out_f_hwc, nullptr, nullptr, out_final, CDIM, CDIM, 0);
}
